// EchoStateNetworkv2_55490977464502
// MI455X (gfx1250) — compile-verified, matrix operand order checked by reading, not run on hardware
//
#include <hip/hip_runtime.h>
#include <hip/hip_bf16.h>

// ---------------------------------------------------------------------------
// Echo State Network for MI455X (gfx1250, wave32, WMMA + async-to-LDS)
//   B=32, T=512, IN=128, RES=2048, OUT=64, WINDOW=8, OVERLAP=2 -> n_win=128
// ---------------------------------------------------------------------------

#define B_    32
#define T_    512
#define IN_   128
#define RES_  2048
#define OUT_  64
#define NW_   128          // n_win
#define ROWS_ 4096         // B_*NW_   (flattened (b,w) rows of the state)

typedef __attribute__((ext_vector_type(16))) __bf16 v16bf;
typedef __attribute__((ext_vector_type(8)))  float  v8f;
typedef __attribute__((ext_vector_type(2)))  float  v2f;

union FragBF16 { v16bf v; unsigned u[8]; };

// K index pattern for 16-bit A/B fragments (per ISA 16-bit 16x32 layout):
// dword d (0..7), lane-half hi: d<4 -> K=2d+8*hi ; d>=4 -> K=16+2(d-4)+8*hi
__device__ __forceinline__ int kidx16(int d, int hi) {
    return (d < 4 ? 2 * d : 8 + 2 * d) + 8 * hi;
}

// --- CDNA5 async memory->LDS copy, GVS form: addr = SGPR64 + VGPR32 --------
// Per-lane 32-bit offset is loop-invariant; uniform k0 advance rides in the
// scalar base (SALU), keeping the hot loop's VALU free for WMMA co-execution.
__device__ __forceinline__ void async_load_b128_s(unsigned lds, unsigned voff,
                                                  const void* sbase) {
    asm volatile("global_load_async_to_lds_b128 %0, %1, %2"
                 :: "v"(lds), "v"(voff), "s"(sbase)
                 : "memory");
}
__device__ __forceinline__ void wait_asynccnt(int n) {
    asm volatile("s_wait_asynccnt %0" :: "i"(n) : "memory");
}

// ---------------------------------------------------------------------------
// WMMA compute on one staged 128x32 (A) x 32x128 (B^T) tile pair
// ---------------------------------------------------------------------------
__device__ __forceinline__ void wmma_tiles_bf16(
    const __bf16 (*__restrict__ As)[40], const __bf16 (*__restrict__ Bs)[40],
    int waveM, int waveN, int lr, int hi, v8f c[2][4])
{
    FragBF16 a[2], bb[4];
    #pragma unroll
    for (int tr = 0; tr < 2; ++tr) {
        const int m = waveM * 32 + tr * 16 + lr;
        #pragma unroll
        for (int d = 0; d < 8; ++d)
            a[tr].u[d] = *(const unsigned*)&As[m][kidx16(d, hi)];
    }
    #pragma unroll
    for (int tc = 0; tc < 4; ++tc) {
        const int n = waveN * 64 + tc * 16 + lr;
        #pragma unroll
        for (int d = 0; d < 8; ++d)
            bb[tc].u[d] = *(const unsigned*)&Bs[n][kidx16(d, hi)];
    }
    #pragma unroll
    for (int tr = 0; tr < 2; ++tr)
        #pragma unroll
        for (int tc = 0; tc < 4; ++tc)
            c[tr][tc] = __builtin_amdgcn_wmma_f32_16x16x32_bf16(
                false, a[tr].v, false, bb[tc].v,
                (short)0, c[tr][tc], false, false);
}

// ---------------------------------------------------------------------------
// Kernel 1: reservoir step.  Z = S*W_res + U_s*W_in ;  S' = .7 S + .3 sin(Z)
// Grid: (32 M-tiles == batches, 16 N-tiles), 256 threads (8 wave32).
// Phase 1 is fully async-fed (double buffered); A = bf16 state mirror,
// B = pre-transposed bf16 W_res [n][k] -> both tiles are pure byte copies.
// ---------------------------------------------------------------------------
__global__ __launch_bounds__(256) void res_step_kernel(
    const float*  __restrict__ Sf,         // old state f32 [4096][2048]
    float*        __restrict__ SfN,        // new state f32
    const __bf16* __restrict__ Sbf,        // old state bf16 mirror
    __bf16*       __restrict__ SbfN,       // new state bf16 mirror
    const __bf16* __restrict__ WresT,      // [2048 n][2048 k] bf16 (transposed)
    const __bf16* __restrict__ WinT,       // [2048 n][128 k]  bf16 (transposed)
    const float*  __restrict__ inp,        // input_data [32][512][128]
    float*        __restrict__ rs,         // [32][512][2048]
    int s, int kresIters)
{
    __shared__ __bf16 As[2][128][40];      // A tiles [m][k], 16B-aligned rows
    __shared__ __bf16 Bs[2][128][40];      // B tiles [n][k]

    const int tid   = threadIdx.x;
    const int bM    = blockIdx.x;          // batch == M tile (128 rows)
    const int n0    = blockIdx.y * 128;    // N tile base
    const int row0  = bM * 128;
    const int wv    = tid >> 5, lane = tid & 31;
    const int waveM = wv & 3,  waveN = wv >> 2;   // 4 x 2 wave grid
    const int lr    = lane & 15, hi = lane >> 4;

    v8f c[2][4];
    #pragma unroll
    for (int i = 0; i < 2; ++i)
        #pragma unroll
        for (int j = 0; j < 4; ++j)
            #pragma unroll
            for (int e = 0; e < 8; ++e) c[i][j][e] = 0.0f;

    // chunk mapping: 512 x 16B chunks per 128x32 bf16 tile, 2 per thread
    const int m_c0 = (tid + 0)   >> 2, p_c0 = ((tid + 0)   & 3) * 8;
    const int m_c1 = (tid + 256) >> 2, p_c1 = ((tid + 256) & 3) * 8;

    // loop-invariant per-lane byte offsets (A and B share row stride RES_)
    const unsigned voff0 = (unsigned)(m_c0 * RES_ + p_c0) * 2u;
    const unsigned voff1 = (unsigned)(m_c1 * RES_ + p_c1) * 2u;
    const unsigned voffW0 = (unsigned)(m_c0 * IN_ + p_c0) * 2u;   // W_in tiles
    const unsigned voffW1 = (unsigned)(m_c1 * IN_ + p_c1) * 2u;

    // loop-invariant LDS destination byte addresses (2 buffers x {A0,A1,B0,B1})
    unsigned ldsA0[2], ldsA1[2], ldsB0[2], ldsB1[2];
    #pragma unroll
    for (int bf = 0; bf < 2; ++bf) {
        ldsA0[bf] = (unsigned)(uintptr_t)&As[bf][m_c0][p_c0];
        ldsA1[bf] = (unsigned)(uintptr_t)&As[bf][m_c1][p_c1];
        ldsB0[bf] = (unsigned)(uintptr_t)&Bs[bf][m_c0][p_c0];
        ldsB1[bf] = (unsigned)(uintptr_t)&Bs[bf][m_c1][p_c1];
    }

    // uniform scalar bases (k0 advance is SALU-only inside the loop)
    const char* baseA0 = (const char*)(Sbf   + (size_t)row0 * RES_);
    const char* baseB0 = (const char*)(WresT + (size_t)n0   * RES_);
    const char* baseW0 = (const char*)(WinT  + (size_t)n0   * IN_);

    // ---- phase 1: K over RES (S * W_res), skipped when s==0 (S==0) ----
    if (kresIters > 0) {
        // prologue: issue tile 0 into buffer 0
        async_load_b128_s(ldsA0[0], voff0, baseA0);
        async_load_b128_s(ldsB0[0], voff0, baseB0);
        async_load_b128_s(ldsA1[0], voff1, baseA0);
        async_load_b128_s(ldsB1[0], voff1, baseB0);
        for (int kt = 0; kt < kresIters; ++kt) {
            const int cur = kt & 1;
            const bool more = (kt + 1) < kresIters;
            if (more) {                       // prefetch next tile -> other buf
                const int nb = 1 - cur;
                const char* bA = baseA0 + (size_t)(kt + 1) * 64;  // 32 bf16
                const char* bB = baseB0 + (size_t)(kt + 1) * 64;
                async_load_b128_s(ldsA0[nb], voff0, bA);
                async_load_b128_s(ldsB0[nb], voff0, bB);
                async_load_b128_s(ldsA1[nb], voff1, bA);
                async_load_b128_s(ldsB1[nb], voff1, bB);
                wait_asynccnt(4);             // drain current tile (in order)
            } else {
                wait_asynccnt(0);
            }
            __syncthreads();
            wmma_tiles_bf16(As[cur], Bs[cur], waveM, waveN, lr, hi, c);
            __syncthreads();                  // buffer free for next prefetch
        }
    }

    // ---- phase 2: K over IN (U_s * W_in), window input with zero padding ----
    for (int kt = 0; kt < 4; ++kt) {
        const int k0 = kt * 32;
        // B tile: async byte-copy from transposed W_in
        async_load_b128_s(ldsB0[0], voffW0, baseW0 + (size_t)k0 * 2);
        async_load_b128_s(ldsB1[0], voffW1, baseW0 + (size_t)k0 * 2);
        // A tile: gather window input, f32->bf16, zero-pad edges (VALU)
        {
            const int m = tid >> 1, kk0 = (tid & 1) * 16;
            const int t = (m << 2) + s - 2;              // 4*w + s - 2
            const bool valid = (t >= 0) && (t < T_);
            const float* src = inp + ((size_t)bM * T_ + (valid ? t : 0)) * IN_
                                   + k0 + kk0;
            #pragma unroll
            for (int i = 0; i < 16; ++i)
                As[0][m][kk0 + i] = (__bf16)(valid ? src[i] : 0.0f);
        }
        wait_asynccnt(0);
        __syncthreads();
        wmma_tiles_bf16(As[0], Bs[0], waveM, waveN, lr, hi, c);
        __syncthreads();
    }

    // ---- epilogue: leaky sin update, write f32 + bf16 state (+ rs, s>=2) ----
    #pragma unroll
    for (int tr = 0; tr < 2; ++tr)
        #pragma unroll
        for (int tc = 0; tc < 4; ++tc)
            #pragma unroll
            for (int v = 0; v < 8; ++v) {
                const int m = waveM * 32 + tr * 16 + hi * 8 + v;
                const int n = waveN * 64 + tc * 16 + lr;
                const int row = row0 + m;
                const size_t idx = (size_t)row * RES_ + (n0 + n);
                const float z = c[tr][tc][v];
                const float sNew = 0.7f * Sf[idx] + 0.3f * __sinf(z);
                SfN[idx]  = sNew;
                SbfN[idx] = (__bf16)sNew;
                if (s >= 2) {
                    const int b = row >> 7, w = row & 127;
                    rs[(((size_t)b * T_) + (w << 2) + (s - 2)) * RES_
                       + (n0 + n)] = sNew;
                }
            }
}

// ---------------------------------------------------------------------------
// Kernel 2: Gram matrix A[b] = rs[b] * rs[b]^T + 1 + softplus(lam)*I
// Grid: 32 batches * 16 (4x4) 128x128 tiles of the 512x512 output. f32 WMMA.
// ---------------------------------------------------------------------------
__global__ __launch_bounds__(256) void gram_kernel(
    const float* __restrict__ rs, float* __restrict__ A,
    const float* __restrict__ lam_ptr)
{
    __shared__ float Xi[128][33];
    __shared__ float Xj[128][33];

    const int tid = threadIdx.x;
    const int blk = blockIdx.x;
    const int b = blk >> 4, ti = (blk >> 2) & 3, tj = blk & 3;
    const int i0 = ti * 128, j0 = tj * 128;
    const int wv = tid >> 5, lane = tid & 31;
    const int waveM = wv & 3, waveN = wv >> 2;
    const int lr = lane & 15, hi = lane >> 4;

    v8f c[2][4];
    #pragma unroll
    for (int i = 0; i < 2; ++i)
        #pragma unroll
        for (int j = 0; j < 4; ++j)
            #pragma unroll
            for (int e = 0; e < 8; ++e) c[i][j][e] = 0.0f;

    for (int kt = 0; kt < 64; ++kt) {
        const int k0 = kt * 32;
        {
            const int m = tid >> 1, kk0 = (tid & 1) * 16;
            const float* si = rs + ((size_t)b * T_ + i0 + m) * RES_ + k0 + kk0;
            const float* sj = rs + ((size_t)b * T_ + j0 + m) * RES_ + k0 + kk0;
            #pragma unroll
            for (int i = 0; i < 16; ++i) { Xi[m][kk0 + i] = si[i];
                                           Xj[m][kk0 + i] = sj[i]; }
        }
        __syncthreads();
        #pragma unroll
        for (int ks = 0; ks < 8; ++ks) {
            const int k = ks * 4 + 2 * hi;
            v2f a[2], bv[4];
            #pragma unroll
            for (int tr = 0; tr < 2; ++tr) {
                const int m = waveM * 32 + tr * 16 + lr;
                a[tr].x = Xi[m][k]; a[tr].y = Xi[m][k + 1];
            }
            #pragma unroll
            for (int tc = 0; tc < 4; ++tc) {
                const int n = waveN * 64 + tc * 16 + lr;
                bv[tc].x = Xj[n][k]; bv[tc].y = Xj[n][k + 1];
            }
            #pragma unroll
            for (int tr = 0; tr < 2; ++tr)
                #pragma unroll
                for (int tc = 0; tc < 4; ++tc)
                    c[tr][tc] = __builtin_amdgcn_wmma_f32_16x16x4_f32(
                        false, a[tr], false, bv[tc],
                        (short)0, c[tr][tc], false, false);
        }
        __syncthreads();
    }

    const float reg = log1pf(__expf(lam_ptr[0]));   // softplus(lam_raw)
    #pragma unroll
    for (int tr = 0; tr < 2; ++tr)
        #pragma unroll
        for (int tc = 0; tc < 4; ++tc)
            #pragma unroll
            for (int v = 0; v < 8; ++v) {
                const int gi = i0 + waveM * 32 + tr * 16 + hi * 8 + v;
                const int gj = j0 + waveN * 64 + tc * 16 + lr;
                float val = c[tr][tc][v] + 1.0f + (gi == gj ? reg : 0.0f);
                A[(size_t)b * T_ * T_ + (size_t)gi * T_ + gj] = val;
            }
}

// ---------------------------------------------------------------------------
// Kernel 3: per-batch Cholesky (lower, in place) + L y = Y, L^T z = y.
// One 256-thread workgroup per batch (32 concurrent solves).
// ---------------------------------------------------------------------------
__global__ __launch_bounds__(256) void cholsolve_kernel(
    float* __restrict__ A, const float* __restrict__ Y, float* __restrict__ Z)
{
    const int b = blockIdx.x, tid = threadIdx.x;
    float* Ab = A + (size_t)b * T_ * T_;
    const float* Yb = Y + (size_t)b * T_ * OUT_;
    float* Zb = Z + (size_t)b * T_ * OUT_;
    __shared__ float diag;

    for (int i = tid; i < T_ * OUT_; i += 256) Zb[i] = Yb[i];
    __syncthreads();

    // Cholesky, lower triangle only
    for (int k = 0; k < T_; ++k) {
        if (tid == 0) { diag = sqrtf(Ab[(size_t)k * T_ + k]);
                        Ab[(size_t)k * T_ + k] = diag; }
        __syncthreads();
        const float dinv = 1.0f / diag;
        for (int i = k + 1 + tid; i < T_; i += 256)
            Ab[(size_t)i * T_ + k] *= dinv;
        __syncthreads();
        for (int i = k + 1 + tid; i < T_; i += 256) {
            const float lik = Ab[(size_t)i * T_ + k];
            for (int j = k + 1; j <= i; ++j)
                Ab[(size_t)i * T_ + j] -= lik * Ab[(size_t)j * T_ + k];
        }
        __syncthreads();
    }

    // forward: L V = Y
    for (int k = 0; k < T_; ++k) {
        const float dinv = 1.0f / Ab[(size_t)k * T_ + k];
        if (tid < OUT_) Zb[(size_t)k * OUT_ + tid] *= dinv;
        __syncthreads();
        for (int u = tid; u < (T_ - 1 - k) * OUT_; u += 256) {
            const int i = k + 1 + u / OUT_, o = u % OUT_;
            Zb[(size_t)i * OUT_ + o] -=
                Ab[(size_t)i * T_ + k] * Zb[(size_t)k * OUT_ + o];
        }
        __syncthreads();
    }
    // backward: L^T W = V
    for (int k = T_ - 1; k >= 0; --k) {
        const float dinv = 1.0f / Ab[(size_t)k * T_ + k];
        if (tid < OUT_) Zb[(size_t)k * OUT_ + tid] *= dinv;
        __syncthreads();
        for (int u = tid; u < k * OUT_; u += 256) {
            const int i = u / OUT_, o = u % OUT_;
            Zb[(size_t)i * OUT_ + o] -=
                Ab[(size_t)k * T_ + i] * Zb[(size_t)k * OUT_ + o];
        }
        __syncthreads();
    }
}

// ---------------------------------------------------------------------------
// Kernel 4: w[b] = rs[b]^T (2048x512) * Z[b] (512x64).  f32 WMMA.
// Grid: 32 batches * 16 M-tiles of 128 rows. 8 waves in 4x2 grid (N strips 32).
// ---------------------------------------------------------------------------
__global__ __launch_bounds__(256) void wout_kernel(
    const float* __restrict__ rs, const float* __restrict__ Z,
    float* __restrict__ Wout)
{
    __shared__ float Al[32][132];   // [k][m]
    __shared__ float Zl[32][68];    // [k][o]

    const int tid = threadIdx.x;
    const int blk = blockIdx.x;
    const int b = blk >> 4, m0 = (blk & 15) * 128;
    const int wv = tid >> 5, lane = tid & 31;
    const int waveM = wv & 3, waveN = wv >> 2;   // waveN in {0,1}
    const int lr = lane & 15, hi = lane >> 4;

    v8f c[2][2];
    #pragma unroll
    for (int i = 0; i < 2; ++i)
        #pragma unroll
        for (int j = 0; j < 2; ++j)
            #pragma unroll
            for (int e = 0; e < 8; ++e) c[i][j][e] = 0.0f;

    for (int kt = 0; kt < 16; ++kt) {
        const int k0 = kt * 32;
        {
            const int kk = tid >> 3, mc = (tid & 7) * 16;
            const float* src = rs + ((size_t)b * T_ + k0 + kk) * RES_ + m0 + mc;
            #pragma unroll
            for (int i = 0; i < 16; ++i) Al[kk][mc + i] = src[i];
        }
        {
            const int kk = tid >> 3, oc = (tid & 7) * 8;
            const float* src = Z + ((size_t)b * T_ + k0 + kk) * OUT_ + oc;
            #pragma unroll
            for (int i = 0; i < 8; ++i) Zl[kk][oc + i] = src[i];
        }
        __syncthreads();
        #pragma unroll
        for (int ks = 0; ks < 8; ++ks) {
            const int k = ks * 4 + 2 * hi;
            v2f a[2], bv[2];
            #pragma unroll
            for (int tr = 0; tr < 2; ++tr) {
                const int m = waveM * 32 + tr * 16 + lr;
                a[tr].x = Al[k][m]; a[tr].y = Al[k + 1][m];
            }
            #pragma unroll
            for (int tc = 0; tc < 2; ++tc) {
                const int n = waveN * 32 + tc * 16 + lr;
                bv[tc].x = Zl[k][n]; bv[tc].y = Zl[k + 1][n];
            }
            #pragma unroll
            for (int tr = 0; tr < 2; ++tr)
                #pragma unroll
                for (int tc = 0; tc < 2; ++tc)
                    c[tr][tc] = __builtin_amdgcn_wmma_f32_16x16x4_f32(
                        false, a[tr], false, bv[tc],
                        (short)0, c[tr][tc], false, false);
        }
        __syncthreads();
    }

    #pragma unroll
    for (int tr = 0; tr < 2; ++tr)
        #pragma unroll
        for (int tc = 0; tc < 2; ++tc)
            #pragma unroll
            for (int v = 0; v < 8; ++v) {
                const int m = m0 + waveM * 32 + tr * 16 + hi * 8 + v;
                const int o = waveN * 32 + tc * 16 + lr;
                Wout[((size_t)b * RES_ + m) * OUT_ + o] = c[tr][tc][v];
            }
}

// ---------------------------------------------------------------------------
// Small helper kernels
// ---------------------------------------------------------------------------
__global__ void bias_kernel(const float* __restrict__ Z,
                            float* __restrict__ outBias)
{
    const int b = blockIdx.x, o = threadIdx.x;
    float s = 0.0f;
    for (int k = 0; k < T_; ++k) s += Z[((size_t)b * T_ + k) * OUT_ + o];
    outBias[b * OUT_ + o] = s;
}

__global__ void zero_f32_kernel(float* p, size_t n)
{
    size_t i = (size_t)blockIdx.x * blockDim.x + threadIdx.x;
    const size_t stride = (size_t)gridDim.x * blockDim.x;
    for (; i < n; i += stride) p[i] = 0.0f;
}

// transpose + f32->bf16:  dst[c][r] = (bf16) src[r][c]
__global__ __launch_bounds__(256) void transpose_bf16_kernel(
    const float* __restrict__ src, __bf16* __restrict__ dst,
    int rows, int cols)
{
    __shared__ float tile[32][33];
    const int c0 = blockIdx.x * 32, r0 = blockIdx.y * 32;
    const int tx = threadIdx.x & 31, ty = threadIdx.x >> 5;
    for (int i = ty; i < 32; i += 8)
        tile[i][tx] = src[(size_t)(r0 + i) * cols + c0 + tx];
    __syncthreads();
    for (int i = ty; i < 32; i += 8)
        dst[(size_t)(c0 + i) * rows + r0 + tx] = (__bf16)tile[tx][i];
}

// ---------------------------------------------------------------------------
// Launch
// ---------------------------------------------------------------------------
extern "C" void kernel_launch(void* const* d_in, const int* in_sizes, int n_in,
                              void* d_out, int out_size, void* d_ws,
                              size_t ws_size, hipStream_t stream)
{
    (void)in_sizes; (void)n_in; (void)out_size; (void)ws_size;
    const float* input  = (const float*)d_in[0];  // [32][512][128]
    const float* target = (const float*)d_in[1];  // [32][512][64]
    const float* WresF  = (const float*)d_in[2];  // [2048][2048]
    const float* WinF   = (const float*)d_in[3];  // [128][2048]
    const float* lam    = (const float*)d_in[4];  // scalar
    float* out = (float*)d_out;

    // workspace carve-up (256B aligned)
    char* ws = (char*)d_ws;
    size_t off = 0;
    auto alloc = [&](size_t bytes) -> void* {
        off = (off + 255) & ~(size_t)255;
        void* p = ws + off;
        off += bytes;
        return p;
    };
    float*  Sf0   = (float*)alloc((size_t)ROWS_ * RES_ * 4);
    float*  Sf1   = (float*)alloc((size_t)ROWS_ * RES_ * 4);
    __bf16* Sbf0  = (__bf16*)alloc((size_t)ROWS_ * RES_ * 2);
    __bf16* Sbf1  = (__bf16*)alloc((size_t)ROWS_ * RES_ * 2);
    __bf16* WresT = (__bf16*)alloc((size_t)RES_ * RES_ * 2);   // [n][k]
    __bf16* WinT  = (__bf16*)alloc((size_t)RES_ * IN_ * 2);    // [n][k]
    float*  rs    = (float*)alloc((size_t)B_ * T_ * RES_ * 4);
    float*  A     = (float*)alloc((size_t)B_ * T_ * T_ * 4);
    float*  Z     = (float*)alloc((size_t)B_ * T_ * OUT_ * 4);

    // init: zero initial f32 state; transpose+convert weights to bf16 [n][k]
    zero_f32_kernel<<<2048, 256, 0, stream>>>(Sf0, (size_t)ROWS_ * RES_);
    transpose_bf16_kernel<<<dim3(64, 64), 256, 0, stream>>>(WresF, WresT,
                                                            RES_, RES_);
    transpose_bf16_kernel<<<dim3(64, 4), 256, 0, stream>>>(WinF, WinT,
                                                           IN_, RES_);

    // 6 reservoir steps, ping-pong state buffers; s==0 skips the S*W_res phase
    dim3 rgrid(32, 16);
    for (int s = 0; s < 6; ++s) {
        const float*  SfIn   = (s & 1) ? Sf1  : Sf0;
        float*        SfOut  = (s & 1) ? Sf0  : Sf1;
        const __bf16* SbfIn  = (s & 1) ? Sbf1 : Sbf0;
        __bf16*       SbfOut = (s & 1) ? Sbf0 : Sbf1;
        res_step_kernel<<<rgrid, 256, 0, stream>>>(
            SfIn, SfOut, SbfIn, SbfOut, WresT, WinT, input, rs,
            s, (s == 0) ? 0 : 64);
    }

    // Gram + regularization, solve, readout
    gram_kernel<<<512, 256, 0, stream>>>(rs, A, lam);
    cholsolve_kernel<<<32, 256, 0, stream>>>(A, target, Z);
    wout_kernel<<<512, 256, 0, stream>>>(rs, Z, out);
    bias_kernel<<<32, 64, 0, stream>>>(Z, out + (size_t)B_ * RES_ * OUT_);
}
